// MHA_44676249813457
// MI455X (gfx1250) — matrix operand order repaired and checked by reading, not compiled, not run on hardware
//
#include <hip/hip_runtime.h>
#include <hip/hip_bf16.h>

typedef __attribute__((ext_vector_type(16))) _Float16 v16h;
typedef __attribute__((ext_vector_type(8)))  float    v8f;

#define WMMA_F16(a, b, c) \
  __builtin_amdgcn_wmma_f32_16x16x32_f16(false, (a), false, (b), (short)0, (c), false, false)

// ---------------- WMMA register-layout helpers (CDNA5 ISA §7.12.2) ----------------
// A 16x32 f16: lane {grp,l16}; row = rbase + l16;
//   elements i -> K = grp*16 + i (contiguous)
// => one contiguous 32-byte run per lane.
__device__ __forceinline__ v16h load_a_f16(const _Float16* base, int ld,
                                           int row, int kk, int lane) {
  const int koff = (lane & 16) ? 16 : 0;
  const _Float16* p0 = base + (size_t)row * ld + kk + koff;
  const _Float16* p1 = p0 + 8;
  v16h a;
#pragma unroll
  for (int i = 0; i < 8; ++i) { a[i] = p0[i]; a[8 + i] = p1[i]; }
  return a;
}

// B 32x16 f16: lane col = l16; elements i -> K = grp*16 + i (contiguous).
// src laid out K-contiguous per column (i.e., W^T row-major), ld = K stride.
__device__ __forceinline__ v16h load_b_kcontig(const _Float16* colbase,
                                               int kk, int lane) {
  const int kb = (lane & 16) ? 16 : 0;
  const _Float16* p = colbase + kk + kb;  // 16 contiguous f16 = 32B
  v16h b;
#pragma unroll
  for (int i = 0; i < 16; ++i) b[i] = p[i];
  return b;
}

// ============================ 0a) x -> f16 ============================
__global__ __launch_bounds__(256) void prep_x(const float* __restrict__ x,
                                              _Float16* __restrict__ xh) {
  const size_t i = ((size_t)blockIdx.x * blockDim.x + threadIdx.x) * 4;
  const float4 v = *(const float4*)(x + i);
  xh[i + 0] = (_Float16)v.x;
  xh[i + 1] = (_Float16)v.y;
  xh[i + 2] = (_Float16)v.z;
  xh[i + 3] = (_Float16)v.w;
}

// ============================ 0b) W -> f16, transposed ============================
// WT[n][k] = W[k][n]; 32x32 tiles through LDS. blockIdx.z selects the matrix.
__global__ __launch_bounds__(256) void prep_wt(
    const float* __restrict__ Wq, const float* __restrict__ Wk,
    const float* __restrict__ Wv, const float* __restrict__ Wo,
    _Float16* __restrict__ WT) {
  const float* src = (blockIdx.z == 0) ? Wq
                   : (blockIdx.z == 1) ? Wk
                   : (blockIdx.z == 2) ? Wv : Wo;
  _Float16* dst = WT + (size_t)blockIdx.z * 1024 * 1024;
  __shared__ _Float16 t[32][33];
  const int tx = threadIdx.x, ty = threadIdx.y;       // 32 x 8
  const int n0 = blockIdx.x * 32, k0 = blockIdx.y * 32;
#pragma unroll
  for (int j = 0; j < 32; j += 8)
    t[ty + j][tx] = (_Float16)src[(size_t)(k0 + ty + j) * 1024 + n0 + tx];
  __syncthreads();
#pragma unroll
  for (int j = 0; j < 32; j += 8)
    dst[(size_t)(n0 + ty + j) * 1024 + k0 + tx] = t[tx][ty + j];
}

// ============================ 1) fused QKV projection ============================
// Each wave: 64x16 output (4 row-tiles) x 3 weights = 12 WMMAs per k-step.
// Q,K head-major [bh, n, 64]; V transposed [bh, 64, n] for contiguous P·V B-loads.
__global__ __launch_bounds__(256) void qkv_kernel(
    const _Float16* __restrict__ xh, const _Float16* __restrict__ WT,
    _Float16* __restrict__ Q, _Float16* __restrict__ K, _Float16* __restrict__ V) {
  const int wave = blockIdx.x * (blockDim.x >> 5) + (threadIdx.x >> 5);
  const int lane = threadIdx.x & 31;
  const int l16 = lane & 15, grp = lane >> 4;
  const int jc = wave & 63;   // col tile (1024/16)
  const int rg = wave >> 6;   // row group (8192/64)
  const int rbase = rg * 64, cbase = jc * 16;

  const _Float16* WqT = WT;
  const _Float16* WkT = WT + (size_t)1024 * 1024;
  const _Float16* WvT = WT + (size_t)2 * 1024 * 1024;
  const _Float16* cq_col = WqT + (size_t)(cbase + l16) * 1024;
  const _Float16* ck_col = WkT + (size_t)(cbase + l16) * 1024;
  const _Float16* cv_col = WvT + (size_t)(cbase + l16) * 1024;

  v8f cq[4] = {}, ck[4] = {}, cv[4] = {};
  for (int kk = 0; kk < 1024; kk += 32) {
    v16h a[4];
#pragma unroll
    for (int i4 = 0; i4 < 4; ++i4)
      a[i4] = load_a_f16(xh, 1024, rbase + i4 * 16 + l16, kk, lane);
    const v16h bq = load_b_kcontig(cq_col, kk, lane);
    const v16h bk = load_b_kcontig(ck_col, kk, lane);
    const v16h bv = load_b_kcontig(cv_col, kk, lane);
#pragma unroll
    for (int i4 = 0; i4 < 4; ++i4) {
      cq[i4] = WMMA_F16(a[i4], bq, cq[i4]);
      ck[i4] = WMMA_F16(a[i4], bk, ck[i4]);
      cv[i4] = WMMA_F16(a[i4], bv, cv[i4]);
    }
  }
  const int cc = cbase + l16;
  const int h = cc >> 6, hd = cc & 63;
#pragma unroll
  for (int i4 = 0; i4 < 4; ++i4) {
#pragma unroll
    for (int r = 0; r < 8; ++r) {
      const int rr = rbase + i4 * 16 + grp * 8 + r;
      const int b = rr >> 11, n = rr & 2047;
      const size_t bh = (size_t)(b * 16 + h);
      Q[(bh * 2048 + n) * 64 + hd] = (_Float16)cq[i4][r];
      K[(bh * 2048 + n) * 64 + hd] = (_Float16)ck[i4][r];
      V[(bh * 64 + hd) * 2048 + n] = (_Float16)cv[i4][r];  // transposed
    }
  }
}

// ============================ 2) causal flash attention ============================
__global__ __launch_bounds__(32) void attn_kernel(
    const _Float16* __restrict__ Q, const _Float16* __restrict__ K,
    const _Float16* __restrict__ V, _Float16* __restrict__ O) {
  __shared__ __align__(16) _Float16 ldsP[16 * 32];  // P tile, row-major 16x32

  const int qt = blockIdx.x & 127;  // query tile (2048/16)
  const int bh = blockIdx.x >> 7;   // b*16 + h
  const int lane = threadIdx.x & 31;
  const int l16 = lane & 15, grp = lane >> 4;

  const _Float16* Qb = Q + (size_t)bh * 2048 * 64;
  const _Float16* Kb = K + (size_t)bh * 2048 * 64;
  const _Float16* Vb = V + (size_t)bh * 2048 * 64;  // [hd][n]

  const int q0 = qt * 16;
  const v16h aq0 = load_a_f16(Qb, 64, q0 + l16, 0, lane);
  const v16h aq1 = load_a_f16(Qb, 64, q0 + l16, 32, lane);

  float m[8], l[8];
  v8f av[4] = {};
#pragma unroll
  for (int r = 0; r < 8; ++r) { m[r] = -1e30f; l[r] = 0.0f; }

  const int kmax = q0 + 16;
  const int kb = grp * 16;
  for (int j0 = 0; j0 < kmax; j0 += 32) {
    // ---- scores: keys j0..j0+15 (tile0), j0+16..j0+31 (tile1) ----
    const int key0 = j0 + l16;
    const int key1 = j0 + 16 + l16;
    const int key1c = key1 > 2047 ? 2047 : key1;  // clamp loads; masked below
    v16h bk0lo, bk0hi, bk1lo, bk1hi;
    {
      const _Float16* p0 = Kb + (size_t)key0 * 64 + kb;
      const _Float16* p1 = Kb + (size_t)key1c * 64 + kb;
#pragma unroll
      for (int i = 0; i < 16; ++i) {
        bk0lo[i] = p0[i];      bk0hi[i] = p0[32 + i];
        bk1lo[i] = p1[i];      bk1hi[i] = p1[32 + i];
      }
    }
    v8f c0 = {}, c1 = {};
    c0 = WMMA_F16(aq0, bk0lo, c0);
    c0 = WMMA_F16(aq1, bk0hi, c0);
    c1 = WMMA_F16(aq0, bk1lo, c1);
    c1 = WMMA_F16(aq1, bk1hi, c1);

    // ---- online softmax (xor 1/2/4/8 stays inside each 16-lane C group) ----
#pragma unroll
    for (int r = 0; r < 8; ++r) {
      const int qg = q0 + grp * 8 + r;
      float s0 = (key0 <= qg) ? c0[r] * 0.125f : -1e30f;  // 1/sqrt(64)
      float s1 = (key1 <= qg) ? c1[r] * 0.125f : -1e30f;
      float mx = fmaxf(s0, s1);
      mx = fmaxf(mx, __shfl_xor(mx, 1));
      mx = fmaxf(mx, __shfl_xor(mx, 2));
      mx = fmaxf(mx, __shfl_xor(mx, 4));
      mx = fmaxf(mx, __shfl_xor(mx, 8));
      const float nm = fmaxf(m[r], mx);
      const float alpha = __expf(m[r] - nm);
      const float p0 = __expf(s0 - nm);
      const float p1 = __expf(s1 - nm);
      float rs = p0 + p1;
      rs += __shfl_xor(rs, 1);
      rs += __shfl_xor(rs, 2);
      rs += __shfl_xor(rs, 4);
      rs += __shfl_xor(rs, 8);
      l[r] = l[r] * alpha + rs;
      m[r] = nm;
      av[0][r] *= alpha; av[1][r] *= alpha;
      av[2][r] *= alpha; av[3][r] *= alpha;
      ldsP[(grp * 8 + r) * 32 + l16]      = (_Float16)p0;
      ldsP[(grp * 8 + r) * 32 + 16 + l16] = (_Float16)p1;
    }
    __syncthreads();  // C-layout -> A-layout transpose via LDS

    v16h ap;
    {
      const _Float16* p = &ldsP[l16 * 32 + grp * 16];
#pragma unroll
      for (int i = 0; i < 8; ++i) { ap[i] = p[i]; ap[8 + i] = p[8 + i]; }
    }
    // ---- av += P(16x32) * V(32x64), V^T => contiguous B loads ----
#pragma unroll
    for (int t = 0; t < 4; ++t) {
      const int colv = t * 16 + l16;
      const _Float16* p = Vb + (size_t)colv * 2048 + j0 + kb;  // may read <=32B
      v16h bv;                                                 // into pad; P==0
#pragma unroll
      for (int i = 0; i < 16; ++i) bv[i] = p[i];
      av[t] = WMMA_F16(ap, bv, av[t]);
    }
    __syncthreads();
  }

  // ---- epilogue: O[b, n, h*64+col] = av / l ----
  const int b = bh >> 4, h = bh & 15;
#pragma unroll
  for (int r = 0; r < 8; ++r) {
    const int rr = q0 + grp * 8 + r;
    const float inv = 1.0f / l[r];
#pragma unroll
    for (int t = 0; t < 4; ++t) {
      const size_t off = ((size_t)b * 2048 + rr) * 1024 + h * 64 + t * 16 + l16;
      O[off] = (_Float16)(av[t][r] * inv);
    }
  }
}

// ============================ 3) output projection + bias ============================
__global__ __launch_bounds__(256) void oproj_kernel(
    const _Float16* __restrict__ O, const _Float16* __restrict__ WoT,
    const float* __restrict__ bo, float* __restrict__ out) {
  const int wave = blockIdx.x * (blockDim.x >> 5) + (threadIdx.x >> 5);
  const int lane = threadIdx.x & 31;
  const int l16 = lane & 15, grp = lane >> 4;
  const int jc = wave & 63;
  const int rg = wave >> 6;
  const int rbase = rg * 64, cbase = jc * 16;
  const _Float16* col = WoT + (size_t)(cbase + l16) * 1024;

  v8f c[4] = {};
  for (int kk = 0; kk < 1024; kk += 32) {
    v16h a[4];
#pragma unroll
    for (int i4 = 0; i4 < 4; ++i4)
      a[i4] = load_a_f16(O, 1024, rbase + i4 * 16 + l16, kk, lane);
    const v16h b = load_b_kcontig(col, kk, lane);
#pragma unroll
    for (int i4 = 0; i4 < 4; ++i4) c[i4] = WMMA_F16(a[i4], b, c[i4]);
  }
  const int cc = cbase + l16;
  const float bias = bo[cc];
#pragma unroll
  for (int i4 = 0; i4 < 4; ++i4)
#pragma unroll
    for (int r = 0; r < 8; ++r) {
      const int rr = rbase + i4 * 16 + grp * 8 + r;
      out[(size_t)rr * 1024 + cc] = c[i4][r] + bias;
    }
}

// ============================ launch ============================
extern "C" void kernel_launch(void* const* d_in, const int* in_sizes, int n_in,
                              void* d_out, int out_size, void* d_ws, size_t ws_size,
                              hipStream_t stream) {
  const float* x  = (const float*)d_in[0];
  const float* Wq = (const float*)d_in[1];
  const float* Wk = (const float*)d_in[2];
  const float* Wv = (const float*)d_in[3];
  const float* Wo = (const float*)d_in[4];
  const float* bo = (const float*)d_in[5];
  float* out = (float*)d_out;

  const size_t SZ = (size_t)4 * 2048 * 1024;   // 8,388,608 elements
  const size_t WSZ = (size_t)1024 * 1024;
  _Float16* ws = (_Float16*)d_ws;
  _Float16* xh = ws;                 // SZ
  _Float16* WT = xh + SZ;            // 4 * WSZ (Wq,Wk,Wv,Wo transposed f16)
  _Float16* Q  = WT + 4 * WSZ;       // SZ
  _Float16* K  = Q + SZ;             // SZ
  _Float16* V  = K + SZ;             // SZ (+64 pad for causal tail reads)
  _Float16* O  = V + SZ + 64;        // SZ   (total ~92 MB f16 scratch)

  prep_x<<<8192, 256, 0, stream>>>(x, xh);
  prep_wt<<<dim3(32, 32, 4), dim3(32, 8), 0, stream>>>(Wq, Wk, Wv, Wo, WT);
  // 8192 waves: 4 row-tiles x 64 col-tiles x 3 weights each
  qkv_kernel<<<1024, 256, 0, stream>>>(xh, WT, Q, K, V);
  attn_kernel<<<8192, 32, 0, stream>>>(Q, K, V, O);
  oproj_kernel<<<1024, 256, 0, stream>>>(O, WT + 3 * WSZ, bo, out);
}
